// PI0Pytorch_23167053595059
// MI455X (gfx1250) — compile-verified, matrix operand order checked by reading, not run on hardware
//
#include <hip/hip_runtime.h>
#include <hip/hip_bf16.h>
#include <math.h>

// ---------------------------------------------------------------------------
// PI0 fused transformer layer for MI455X (gfx1250, wave32, WMMA).
// Compute-bound (~730 GFLOP vs ~0.5 GB weight traffic) => all GEMMs
// (QKV, scores, ctx, O, gate/up, down) run on v_wmma_f32_16x16x32_bf16.
// fp32 weights are converted to bf16 during the global->LDS staging pass
// (non-temporal loads: weights are streamed once, keep L2 for activations),
// fully unrolled/vectorized (b128 LDS stores) and double-buffered so global
// loads overlap the WMMA pipeline. Transcendentals use the hardware
// v_exp_f32 path instead of branchy libm expansions.
// ---------------------------------------------------------------------------

typedef __attribute__((ext_vector_type(16))) __bf16 v16bf;
typedef __attribute__((ext_vector_type(8)))  float  v8f;

constexpr int BB = 4;          // batch
constexpr int SV = 768, SE = 64, STOT = 832;
constexpr int DV = 2048, DE = 1024;
constexpr int NH = 8, HD = 256;
constexpr int FV = 16384, FE = 4096;

union FragU { v16bf v; uint4 u[2]; };

static __device__ inline v8f vzero8() {
  v8f z; for (int i = 0; i < 8; ++i) z[i] = 0.0f; return z;
}

// branch-free tanh via hardware exp: tanh(x) = 1 - 2/(e^{2x}+1)
static __device__ inline float fast_tanh(float x) {
  return 1.0f - 2.0f / (__expf(2.0f * x) + 1.0f);
}

// ---------------------------------------------------------------------------
// Generic batched GEMM: D[M,N] (f32) = A[M,K](bf16) x B + optional residual R.
// BTYPE: 0 = B is f32 [K,N] (weights, converted to bf16 on load; N%128==0)
//        1 = B is bf16 [K,N] (N%128==0)
//        2 = B is bf16 [N,K] (B^T; used for Q.K^T; N guarded)
// Block tile 128x128, 8 waves (2 along M x 4 along N), wave tile 64x32.
// K must be a multiple of 32 (true for every GEMM in this layer).
// Double-buffered LDS staging.
// ---------------------------------------------------------------------------
template <int BTYPE>
__global__ __launch_bounds__(256) void gemm_wmma(
    const __bf16* __restrict__ A, const void* __restrict__ Bv,
    const float* __restrict__ R, float* __restrict__ D,
    int M, int N, int K, int lda, int ldb, int ldc,
    long long abatch, long long bbatch, long long cbatch, int bdiv) {
  constexpr int ST = 40;  // LDS row stride in halfs (32 + 8 pad), 80B (16B aligned)
  __shared__ __bf16 As[2][128 * ST];
  __shared__ __bf16 Bs[2][128 * ST];

  const int z = blockIdx.z;
  A += (long long)z * abatch;
  const float*  Bf = (const float*)Bv  + (BTYPE == 0 ? (long long)(z / bdiv) * bbatch : 0);
  const __bf16* Bh = (const __bf16*)Bv + (BTYPE != 0 ? (long long)(z / bdiv) * bbatch : 0);
  D += (long long)z * cbatch;
  if (R) R += (long long)z * cbatch;

  const int tid  = threadIdx.x;
  const int wave = tid >> 5, lane = tid & 31;
  const int mw = wave & 1;        // 0..1  -> 64-row strip
  const int nw = wave >> 1;       // 0..3  -> 32-col strip
  const int m0 = blockIdx.y * 128;
  const int n0 = blockIdx.x * 128;

  v8f acc[4][2];
  for (int i = 0; i < 4; ++i) for (int j = 0; j < 2; ++j) acc[i][j] = vzero8();

  const int r  = lane & 15;
  const int kb = (lane >> 4) * 8;     // half offset within the 32-K slab

  // ---- staging register state + helpers -----------------------------------
  const int arow = tid >> 1, aseg = tid & 1;   // A: 2 threads/row, 16 halfs each
  uint4 ra0, ra1;                               // A staged regs
  uint4 rb0, rb1;                               // B staged regs (BTYPE 2)
  float rf[16];                                 // B staged regs (BTYPE 0/1)
  const int bn  = tid & 127;                    // B: column owner (BTYPE 0/1)
  const int bks = (tid >> 7) * 16;              // 16 consecutive k per thread

  auto loadA = [&](int k0) {
    ra0 = uint4{0, 0, 0, 0}; ra1 = uint4{0, 0, 0, 0};
    int gr = m0 + arow;
    if (gr < M) {
      const uint4* p = (const uint4*)&A[(long long)gr * lda + k0 + aseg * 16];
      ra0 = p[0]; ra1 = p[1];
    }
  };
  auto storeA = [&](int buf) {
    *(uint4*)&As[buf][arow * ST + aseg * 16]     = ra0;
    *(uint4*)&As[buf][arow * ST + aseg * 16 + 8] = ra1;
  };
  auto loadB = [&](int k0) {
    if constexpr (BTYPE == 2) {
      rb0 = uint4{0, 0, 0, 0}; rb1 = uint4{0, 0, 0, 0};
      int gn = n0 + arow;
      if (gn < N) {
        const uint4* p = (const uint4*)&Bh[(long long)gn * ldb + k0 + aseg * 16];
        rb0 = p[0]; rb1 = p[1];
      }
    } else {
      const long long base = (long long)(k0 + bks) * ldb + n0 + bn;
#pragma unroll
      for (int j = 0; j < 16; ++j) {
        if constexpr (BTYPE == 0)
          rf[j] = __builtin_nontemporal_load(&Bf[base + (long long)j * ldb]);
        else
          rf[j] = (float)Bh[base + (long long)j * ldb];
      }
    }
  };
  auto storeB = [&](int buf) {
    if constexpr (BTYPE == 2) {
      *(uint4*)&Bs[buf][arow * ST + aseg * 16]     = rb0;
      *(uint4*)&Bs[buf][arow * ST + aseg * 16 + 8] = rb1;
    } else {
      union { __bf16 h[8]; uint4 u; } p0, p1;
#pragma unroll
      for (int j = 0; j < 8; ++j) { p0.h[j] = (__bf16)rf[j]; p1.h[j] = (__bf16)rf[j + 8]; }
      *(uint4*)&Bs[buf][bn * ST + bks]     = p0.u;
      *(uint4*)&Bs[buf][bn * ST + bks + 8] = p1.u;
    }
  };

  // ---- prologue: stage slab 0 into buffer 0 --------------------------------
  loadA(0); loadB(0);
  storeA(0); storeB(0);

  const int nk = K >> 5;
  for (int it = 0; it < nk; ++it) {
    const int buf = it & 1;
    __syncthreads();

    // fragments from current buffer
    FragU af[4], bf2[2];
#pragma unroll
    for (int mi = 0; mi < 4; ++mi) {
      int base = (mw * 64 + mi * 16 + r) * ST + kb;
      af[mi].u[0] = *(const uint4*)&As[buf][base];
      af[mi].u[1] = *(const uint4*)&As[buf][base + 16];
    }
#pragma unroll
    for (int ni = 0; ni < 2; ++ni) {
      int base = (nw * 32 + ni * 16 + r) * ST + kb;
      bf2[ni].u[0] = *(const uint4*)&Bs[buf][base];
      bf2[ni].u[1] = *(const uint4*)&Bs[buf][base + 16];
    }

    // prefetch next slab into registers (overlaps with WMMA below)
    if (it + 1 < nk) { loadA((it + 1) << 5); loadB((it + 1) << 5); }

#pragma unroll
    for (int mi = 0; mi < 4; ++mi)
#pragma unroll
      for (int ni = 0; ni < 2; ++ni)
        acc[mi][ni] = __builtin_amdgcn_wmma_f32_16x16x32_bf16(
            false, af[mi].v, false, bf2[ni].v, (short)0, acc[mi][ni], false, false);

    if (it + 1 < nk) { storeA(buf ^ 1); storeB(buf ^ 1); }
  }

  // ---- epilogue (f32 C/D layout: lanes 0-15 -> M=rr, lanes 16-31 -> M=8+rr)
  const int r0 = (lane >> 4) * 8;
  const int cn = lane & 15;
#pragma unroll
  for (int mi = 0; mi < 4; ++mi)
#pragma unroll
    for (int ni = 0; ni < 2; ++ni) {
      int gn = n0 + nw * 32 + ni * 16 + cn;
      if (gn >= N) continue;
      int mb = m0 + mw * 64 + mi * 16 + r0;
#pragma unroll
      for (int rr = 0; rr < 8; ++rr) {
        int gm = mb + rr;
        if (gm < M) {
          float v = acc[mi][ni][rr];
          if (R) v += R[(long long)gm * ldc + gn];
          D[(long long)gm * ldc + gn] = v;
        }
      }
    }
}

// ---------------------------------------------------------------------------
// Fused MLP dual GEMM: act[M,N](bf16) = gelu_tanh(A@G) * (A@U).
// Block tile 128x64, 8 waves (4 along M x 2 along N), wave tile 32x32.
// M%128==0, N%64==0, K%32==0 for all call sites -> no guards.
// Double-buffered LDS staging, vectorized f32->bf16 weight conversion.
// ---------------------------------------------------------------------------
__global__ __launch_bounds__(256) void mlp_dual_wmma(
    const __bf16* __restrict__ A, const float* __restrict__ G,
    const float* __restrict__ U, __bf16* __restrict__ Dact, int K, int N) {
  constexpr int ST = 40;
  __shared__ __bf16 As[2][128 * ST];
  __shared__ __bf16 Gs[2][64 * ST];
  __shared__ __bf16 Us[2][64 * ST];

  const int tid = threadIdx.x;
  const int wave = tid >> 5, lane = tid & 31;
  const int mw = wave >> 1;   // 0..3
  const int nw = wave & 1;    // 0..1
  const int m0 = blockIdx.y * 128;
  const int n0 = blockIdx.x * 64;

  v8f accg[2][2], accu[2][2];
  for (int i = 0; i < 2; ++i)
    for (int j = 0; j < 2; ++j) { accg[i][j] = vzero8(); accu[i][j] = vzero8(); }

  const int r = lane & 15;
  const int kb = (lane >> 4) * 8;

  const int arow = tid >> 1, aseg = tid & 1;
  uint4 ra0, ra1;
  float rg[8], ru[8];
  const int bn  = tid & 63;            // weight column owner
  const int bks = (tid >> 6) * 8;      // 8 consecutive k per thread

  auto loadA = [&](int k0) {
    const uint4* p = (const uint4*)&A[(long long)(m0 + arow) * K + k0 + aseg * 16];
    ra0 = p[0]; ra1 = p[1];
  };
  auto storeA = [&](int buf) {
    *(uint4*)&As[buf][arow * ST + aseg * 16]     = ra0;
    *(uint4*)&As[buf][arow * ST + aseg * 16 + 8] = ra1;
  };
  auto loadGU = [&](int k0) {
    const long long base = (long long)(k0 + bks) * N + n0 + bn;
#pragma unroll
    for (int j = 0; j < 8; ++j) {
      rg[j] = __builtin_nontemporal_load(&G[base + (long long)j * N]);
      ru[j] = __builtin_nontemporal_load(&U[base + (long long)j * N]);
    }
  };
  auto storeGU = [&](int buf) {
    union { __bf16 h[8]; uint4 u; } pg, pu;
#pragma unroll
    for (int j = 0; j < 8; ++j) { pg.h[j] = (__bf16)rg[j]; pu.h[j] = (__bf16)ru[j]; }
    *(uint4*)&Gs[buf][bn * ST + bks] = pg.u;
    *(uint4*)&Us[buf][bn * ST + bks] = pu.u;
  };

  loadA(0); loadGU(0);
  storeA(0); storeGU(0);

  const int nk = K >> 5;
  for (int it = 0; it < nk; ++it) {
    const int buf = it & 1;
    __syncthreads();

    FragU af[2], gf[2], uf[2];
#pragma unroll
    for (int mi = 0; mi < 2; ++mi) {
      int base = (mw * 32 + mi * 16 + r) * ST + kb;
      af[mi].u[0] = *(const uint4*)&As[buf][base];
      af[mi].u[1] = *(const uint4*)&As[buf][base + 16];
    }
#pragma unroll
    for (int ni = 0; ni < 2; ++ni) {
      int base = (nw * 32 + ni * 16 + r) * ST + kb;
      gf[ni].u[0] = *(const uint4*)&Gs[buf][base];
      gf[ni].u[1] = *(const uint4*)&Gs[buf][base + 16];
      uf[ni].u[0] = *(const uint4*)&Us[buf][base];
      uf[ni].u[1] = *(const uint4*)&Us[buf][base + 16];
    }

    if (it + 1 < nk) { loadA((it + 1) << 5); loadGU((it + 1) << 5); }

#pragma unroll
    for (int mi = 0; mi < 2; ++mi)
#pragma unroll
      for (int ni = 0; ni < 2; ++ni) {
        accg[mi][ni] = __builtin_amdgcn_wmma_f32_16x16x32_bf16(
            false, af[mi].v, false, gf[ni].v, (short)0, accg[mi][ni], false, false);
        accu[mi][ni] = __builtin_amdgcn_wmma_f32_16x16x32_bf16(
            false, af[mi].v, false, uf[ni].v, (short)0, accu[mi][ni], false, false);
      }

    if (it + 1 < nk) { storeA(buf ^ 1); storeGU(buf ^ 1); }
  }

  const int r0 = (lane >> 4) * 8;
  const int cn = lane & 15;
#pragma unroll
  for (int mi = 0; mi < 2; ++mi)
#pragma unroll
    for (int ni = 0; ni < 2; ++ni) {
      int gn = n0 + nw * 32 + ni * 16 + cn;
      int mb = m0 + mw * 32 + mi * 16 + r0;
#pragma unroll
      for (int rr = 0; rr < 8; ++rr) {
        float g = accg[mi][ni][rr];
        float t = 0.7978845608028654f * (g + 0.044715f * g * g * g);
        float act = 0.5f * g * (1.0f + fast_tanh(t));
        Dact[(long long)(mb + rr) * N + gn] = (__bf16)(act * accu[mi][ni][rr]);
      }
    }
}

// ---------------------------------------------------------------------------
// RMSNorm (Gemma style): out = x * rsqrt(mean(x^2)+eps) * (1+w), cast to bf16.
// One 256-thread block per row.
// ---------------------------------------------------------------------------
__global__ __launch_bounds__(256) void rmsnorm_bf16(
    const float* __restrict__ x, const float* __restrict__ w,
    __bf16* __restrict__ out, int D) {
  const int row = blockIdx.x;
  const float* xr = x + (long long)row * D;
  float s = 0.0f;
  for (int i = threadIdx.x; i < D; i += 256) { float v = xr[i]; s += v * v; }
  for (int o = 16; o > 0; o >>= 1) s += __shfl_xor(s, o, 32);
  __shared__ float red[8];
  __shared__ float rsv;
  int wave = threadIdx.x >> 5, lane = threadIdx.x & 31;
  if (lane == 0) red[wave] = s;
  __syncthreads();
  if (threadIdx.x == 0) {
    float t = 0.0f;
    for (int i = 0; i < 8; ++i) t += red[i];
    rsv = rsqrtf(t / (float)D + 1e-6f);
  }
  __syncthreads();
  float rs = rsv;
  for (int i = threadIdx.x; i < D; i += 256)
    out[(long long)row * D + i] = (__bf16)(xr[i] * rs * (1.0f + w[i]));
}

// ---------------------------------------------------------------------------
// RoPE on Q; output layout [B, H, S, HD] bf16. One thread per rotation pair.
// inv_freq = 10000^(-d2/128) = exp2(-d2 * log2(10000)/128)
// ---------------------------------------------------------------------------
constexpr float ROPE_L2 = 0.10381025296522976f;  // log2(10000)/128

__global__ __launch_bounds__(256) void rope_q_kernel(
    const float* __restrict__ qv, const float* __restrict__ qe,
    const int* __restrict__ pos, __bf16* __restrict__ out) {
  long long t = (long long)blockIdx.x * 256 + threadIdx.x;
  if (t >= (long long)BB * NH * STOT * 128) return;
  int d2 = (int)(t & 127);
  long long u = t >> 7;
  int s = (int)(u % STOT); u /= STOT;
  int h = (int)(u % NH);
  int b = (int)(u / NH);
  int p = pos[b * STOT + s];
  float inv = __builtin_amdgcn_exp2f(-(float)d2 * ROPE_L2);
  float ang = (float)p * inv;
  float c = __cosf(ang), sn = __sinf(ang);
  const float* src = (s < SV)
      ? &qv[((long long)(b * SV + s)) * (NH * HD)]
      : &qe[((long long)(b * SE + (s - SV))) * (NH * HD)];
  float x1 = src[h * HD + d2], x2 = src[h * HD + d2 + 128];
  long long ob = ((long long)((b * NH + h) * STOT + s)) * HD;
  out[ob + d2]       = (__bf16)(x1 * c - x2 * sn);
  out[ob + d2 + 128] = (__bf16)(x2 * c + x1 * sn);
}

// RoPE on K (single MQA head) + cast V. Outputs [B, S, HD] bf16.
__global__ __launch_bounds__(256) void rope_kv_kernel(
    const float* __restrict__ kv, const float* __restrict__ ke,
    const float* __restrict__ vv, const float* __restrict__ ve,
    const int* __restrict__ pos, __bf16* __restrict__ kout,
    __bf16* __restrict__ vout) {
  long long t = (long long)blockIdx.x * 256 + threadIdx.x;
  if (t >= (long long)BB * STOT * 128) return;
  int d2 = (int)(t & 127);
  long long u = t >> 7;
  int s = (int)(u % STOT);
  int b = (int)(u / STOT);
  int p = pos[b * STOT + s];
  float inv = __builtin_amdgcn_exp2f(-(float)d2 * ROPE_L2);
  float ang = (float)p * inv;
  float c = __cosf(ang), sn = __sinf(ang);
  long long src = (s < SV) ? (long long)(b * SV + s) * HD
                           : (long long)(b * SE + (s - SV)) * HD;
  const float* ks = (s < SV) ? &kv[src] : &ke[src];
  const float* vs = (s < SV) ? &vv[src] : &ve[src];
  float k1 = ks[d2], k2 = ks[d2 + 128];
  long long ob = ((long long)(b * STOT + s)) * HD;
  kout[ob + d2]       = (__bf16)(k1 * c - k2 * sn);
  kout[ob + d2 + 128] = (__bf16)(k2 * c + k1 * sn);
  vout[ob + d2]       = (__bf16)vs[d2];
  vout[ob + d2 + 128] = (__bf16)vs[d2 + 128];
}

// ---------------------------------------------------------------------------
// Masked softmax over last dim (S=832). One block per (b,h,q) row.
// probs bf16 out. scale = HD^-0.5 = 1/16.
// ---------------------------------------------------------------------------
__global__ __launch_bounds__(256) void softmax_kernel(
    const float* __restrict__ scores, const unsigned char* __restrict__ mask,
    __bf16* __restrict__ probs) {
  const int row = blockIdx.x;              // b*H*S + h*S + q
  const int q = row % STOT;
  const int b = row / (NH * STOT);
  const float* src = scores + (long long)row * STOT;
  const unsigned char* m = mask + ((long long)b * STOT + q) * STOT;
  __bf16* dst = probs + (long long)row * STOT;

  float vals[4];
  float mx = -3.0e38f;
  for (int j = 0; j < 4; ++j) {
    int k = threadIdx.x + j * 256;
    float v = -1.0e30f;
    if (k < STOT && m[k]) v = src[k] * 0.0625f;
    vals[j] = v;
    if (k < STOT) mx = fmaxf(mx, v);
  }
  for (int o = 16; o > 0; o >>= 1) mx = fmaxf(mx, __shfl_xor(mx, o, 32));
  __shared__ float red[8];
  __shared__ float bmax, bsum;
  int wave = threadIdx.x >> 5, lane = threadIdx.x & 31;
  if (lane == 0) red[wave] = mx;
  __syncthreads();
  if (threadIdx.x == 0) {
    float t = red[0];
    for (int i = 1; i < 8; ++i) t = fmaxf(t, red[i]);
    bmax = t;
  }
  __syncthreads();
  float gm = bmax;
  float s = 0.0f;
  for (int j = 0; j < 4; ++j) {
    int k = threadIdx.x + j * 256;
    if (k < STOT) { vals[j] = __expf(vals[j] - gm); s += vals[j]; }
  }
  for (int o = 16; o > 0; o >>= 1) s += __shfl_xor(s, o, 32);
  if (lane == 0) red[wave] = s;
  __syncthreads();
  if (threadIdx.x == 0) {
    float t = 0.0f;
    for (int i = 0; i < 8; ++i) t += red[i];
    bsum = t;
  }
  __syncthreads();
  float inv = 1.0f / bsum;
  for (int j = 0; j < 4; ++j) {
    int k = threadIdx.x + j * 256;
    if (k < STOT) dst[k] = (__bf16)(vals[j] * inv);
  }
}

// ctx [B,H,S,HD] f32 -> [B,S,H*HD] bf16
__global__ __launch_bounds__(256) void pack_ctx_kernel(
    const float* __restrict__ ctxf, __bf16* __restrict__ ctxb) {
  long long t = (long long)blockIdx.x * 256 + threadIdx.x;
  if (t >= (long long)BB * NH * STOT * HD) return;
  int d = (int)(t & (HD - 1));
  long long u = t >> 8;
  int s = (int)(u % STOT); u /= STOT;
  int h = (int)(u % NH);
  int b = (int)(u / NH);
  ctxb[((long long)(b * STOT + s)) * (NH * HD) + h * HD + d] = (__bf16)ctxf[t];
}

// ---------------------------------------------------------------------------
extern "C" void kernel_launch(void* const* d_in, const int* in_sizes, int n_in,
                              void* d_out, int out_size, void* d_ws, size_t ws_size,
                              hipStream_t stream) {
  const float* vlm_embeds    = (const float*)d_in[0];
  const float* expert_embeds = (const float*)d_in[1];
  const int*   position_ids  = (const int*)d_in[2];
  const unsigned char* attn_mask = (const unsigned char*)d_in[3];
  const float* vlm_ln1_w  = (const float*)d_in[4];
  const float* vlm_q_w    = (const float*)d_in[5];
  const float* vlm_k_w    = (const float*)d_in[6];
  const float* vlm_v_w    = (const float*)d_in[7];
  const float* vlm_o_w    = (const float*)d_in[8];
  const float* vlm_ln2_w  = (const float*)d_in[9];
  const float* vlm_gate_w = (const float*)d_in[10];
  const float* vlm_up_w   = (const float*)d_in[11];
  const float* vlm_down_w = (const float*)d_in[12];
  const float* exp_ln1_w  = (const float*)d_in[13];
  const float* exp_q_w    = (const float*)d_in[14];
  const float* exp_k_w    = (const float*)d_in[15];
  const float* exp_v_w    = (const float*)d_in[16];
  const float* exp_o_w    = (const float*)d_in[17];
  const float* exp_ln2_w  = (const float*)d_in[18];
  const float* exp_gate_w = (const float*)d_in[19];
  const float* exp_up_w   = (const float*)d_in[20];
  const float* exp_down_w = (const float*)d_in[21];

  char* ws = (char*)d_ws;
  size_t off = 0;
  auto alloc = [&](size_t bytes) -> void* {
    size_t a = (off + 255) & ~(size_t)255;
    off = a + bytes;
    return (void*)(ws + a);
  };

  const long long MV = (long long)BB * SV;   // 3072
  const long long ME = (long long)BB * SE;   // 256

  __bf16* vh    = (__bf16*)alloc(MV * DV * 2);
  __bf16* eh    = (__bf16*)alloc(ME * DE * 2);
  float*  q_vlm = (float*) alloc(MV * (NH * HD) * 4);
  float*  q_exp = (float*) alloc(ME * (NH * HD) * 4);
  float*  k_vlm = (float*) alloc(MV * HD * 4);
  float*  k_exp = (float*) alloc(ME * HD * 4);
  float*  v_vlm = (float*) alloc(MV * HD * 4);
  float*  v_exp = (float*) alloc(ME * HD * 4);
  __bf16* qh    = (__bf16*)alloc((long long)BB * NH * STOT * HD * 2);
  __bf16* kh    = (__bf16*)alloc((long long)BB * STOT * HD * 2);
  __bf16* vhh   = (__bf16*)alloc((long long)BB * STOT * HD * 2);
  float*  scores= (float*) alloc((long long)BB * NH * STOT * STOT * 4);
  __bf16* probs = (__bf16*)alloc((long long)BB * NH * STOT * STOT * 2);
  float*  ctxf  = (float*) alloc((long long)BB * NH * STOT * HD * 4);
  __bf16* ctxb  = (__bf16*)alloc((long long)BB * STOT * (NH * HD) * 2);
  float*  hv    = (float*) alloc(MV * DV * 4);
  float*  he    = (float*) alloc(ME * DE * 4);
  __bf16* hnv   = (__bf16*)alloc(MV * DV * 2);
  __bf16* hne   = (__bf16*)alloc(ME * DE * 2);
  __bf16* act_v = (__bf16*)alloc(MV * FV * 2);
  __bf16* act_e = (__bf16*)alloc(ME * FE * 2);
  (void)ws_size; (void)n_in; (void)in_sizes; (void)out_size;

  float* out_v = (float*)d_out;
  float* out_e = out_v + MV * DV;

  auto g128 = [](long long M, long long N, int z) {
    return dim3((unsigned)((N + 127) / 128), (unsigned)((M + 127) / 128), (unsigned)z);
  };

  // 1) input RMSNorm -> bf16 activations
  rmsnorm_bf16<<<(unsigned)MV, 256, 0, stream>>>(vlm_embeds, vlm_ln1_w, vh, DV);
  rmsnorm_bf16<<<(unsigned)ME, 256, 0, stream>>>(expert_embeds, exp_ln1_w, eh, DE);

  // 2) QKV projections (weights f32 [K,N], converted on load)
  gemm_wmma<0><<<g128(MV, NH * HD, 1), 256, 0, stream>>>(
      vh, vlm_q_w, nullptr, q_vlm, (int)MV, NH * HD, DV, DV, NH * HD, NH * HD, 0, 0, 0, 1);
  gemm_wmma<0><<<g128(MV, HD, 1), 256, 0, stream>>>(
      vh, vlm_k_w, nullptr, k_vlm, (int)MV, HD, DV, DV, HD, HD, 0, 0, 0, 1);
  gemm_wmma<0><<<g128(MV, HD, 1), 256, 0, stream>>>(
      vh, vlm_v_w, nullptr, v_vlm, (int)MV, HD, DV, DV, HD, HD, 0, 0, 0, 1);
  gemm_wmma<0><<<g128(ME, NH * HD, 1), 256, 0, stream>>>(
      eh, exp_q_w, nullptr, q_exp, (int)ME, NH * HD, DE, DE, NH * HD, NH * HD, 0, 0, 0, 1);
  gemm_wmma<0><<<g128(ME, HD, 1), 256, 0, stream>>>(
      eh, exp_k_w, nullptr, k_exp, (int)ME, HD, DE, DE, HD, HD, 0, 0, 0, 1);
  gemm_wmma<0><<<g128(ME, HD, 1), 256, 0, stream>>>(
      eh, exp_v_w, nullptr, v_exp, (int)ME, HD, DE, DE, HD, HD, 0, 0, 0, 1);

  // 3) RoPE + pack to WMMA-friendly layouts
  {
    long long tq = (long long)BB * NH * STOT * 128;
    rope_q_kernel<<<(unsigned)((tq + 255) / 256), 256, 0, stream>>>(q_vlm, q_exp, position_ids, qh);
    long long tk = (long long)BB * STOT * 128;
    rope_kv_kernel<<<(unsigned)((tk + 255) / 256), 256, 0, stream>>>(
        k_vlm, k_exp, v_vlm, v_exp, position_ids, kh, vhh);
  }

  // 4) scores = Q x K^T  (NT, bf16), batched over B*H; K shared per b (bdiv=H)
  gemm_wmma<2><<<g128(STOT, STOT, BB * NH), 256, 0, stream>>>(
      qh, kh, nullptr, scores, STOT, STOT, HD, HD, HD, STOT,
      (long long)STOT * HD, (long long)STOT * HD, (long long)STOT * STOT, NH);

  // 5) masked softmax -> bf16 probs
  softmax_kernel<<<(unsigned)(BB * NH * STOT), 256, 0, stream>>>(scores, attn_mask, probs);

  // 6) ctx = P x V (NN, bf16 B), batched over B*H
  gemm_wmma<1><<<g128(STOT, HD, BB * NH), 256, 0, stream>>>(
      probs, vhh, nullptr, ctxf, STOT, HD, STOT, STOT, HD, HD,
      (long long)STOT * STOT, (long long)STOT * HD, (long long)STOT * HD, NH);

  // 7) [B,H,S,HD] -> [B,S,H*HD] bf16
  {
    long long tc = (long long)BB * NH * STOT * HD;
    pack_ctx_kernel<<<(unsigned)((tc + 255) / 256), 256, 0, stream>>>(ctxf, ctxb);
  }

  // 8) O projection with fused residual: h = embeds + ctx @ o_w  (batched over B)
  gemm_wmma<0><<<g128(SV, DV, BB), 256, 0, stream>>>(
      ctxb, vlm_o_w, vlm_embeds, hv, SV, DV, NH * HD, NH * HD, DV, DV,
      (long long)STOT * NH * HD, 0, (long long)SV * DV, 1);
  gemm_wmma<0><<<g128(SE, DE, BB), 256, 0, stream>>>(
      ctxb + (long long)SV * NH * HD, exp_o_w, expert_embeds, he, SE, DE, NH * HD,
      NH * HD, DE, DE, (long long)STOT * NH * HD, 0, (long long)SE * DE, 1);

  // 9) post-attention RMSNorm
  rmsnorm_bf16<<<(unsigned)MV, 256, 0, stream>>>(hv, vlm_ln2_w, hnv, DV);
  rmsnorm_bf16<<<(unsigned)ME, 256, 0, stream>>>(he, exp_ln2_w, hne, DE);

  // 10) fused gate/up dual GEMM + gelu-gate
  mlp_dual_wmma<<<dim3(FV / 64, (unsigned)(MV / 128)), 256, 0, stream>>>(
      hnv, vlm_gate_w, vlm_up_w, act_v, DV, FV);
  mlp_dual_wmma<<<dim3(FE / 64, (unsigned)(ME / 128)), 256, 0, stream>>>(
      hne, exp_gate_w, exp_up_w, act_e, DE, FE);

  // 11) down projection with fused residual -> final outputs
  gemm_wmma<0><<<g128(MV, DV, 1), 256, 0, stream>>>(
      act_v, vlm_down_w, hv, out_v, (int)MV, DV, FV, FV, DV, DV, 0, 0, 0, 1);
  gemm_wmma<0><<<g128(ME, DE, 1), 256, 0, stream>>>(
      act_e, exp_down_w, he, out_e, (int)ME, DE, FE, FE, DE, DE, 0, 0, 0, 1);
}
